// E3Gnn_63883343561092
// MI455X (gfx1250) — matrix operand order repaired and checked by reading, not compiled, not run on hardware
//
#include <hip/hip_runtime.h>
#include <hip/hip_bf16.h>
#include <math.h>

typedef __attribute__((ext_vector_type(16))) _Float16 v16h;
typedef __attribute__((ext_vector_type(8)))  _Float16 v8h;
typedef __attribute__((ext_vector_type(8)))  float    v8f;

#define N_NODES 256
#define NV 16
#define HF 128
#define MU 128
#define L_LAYERS 3
#define B_BATCH 2
#define OUT_HD 64
#define NODES_TOTAL (B_BATCH * N_NODES) /* 512 */

#define ACT_NONE 0
#define ACT_SILU 1
#define ACT_SIG  2

// fast sigmoid: v_exp + v_rcp (approx, ~1ulp rcp)
__device__ __forceinline__ float sigmoid_f(float x) {
    return __builtin_amdgcn_rcpf(1.0f + __expf(-x));
}

// ---------------------------------------------------------------------------
// A-fragment: 16x32 f16 tile from LDS (row-major, `astride` halfs per row).
// Lane l: M = l%16 ; halfs 0..7 -> K = kc*32 + (l/16)*8 + i ; halfs 8..15 -> +16.
// Both runs are 8 contiguous f16 = one 16B LDS load each.
// ---------------------------------------------------------------------------
__device__ __forceinline__ v16h load_a_frag(const _Float16* a, int astride, int lane, int kc) {
    int row = lane & 15;
    int k0  = kc * 32 + ((lane >> 4) << 3);
    const v8h* p0 = (const v8h*)(a + row * astride + k0);
    const v8h* p1 = (const v8h*)(a + row * astride + k0 + 16);
    v8h lo = *p0, hi = *p1;
    v16h out;
#pragma unroll
    for (int i = 0; i < 8; ++i) { out[i] = lo[i]; out[i + 8] = hi[i]; }
    return out;
}

// ---------------------------------------------------------------------------
// Wave GEMM: D[16 x NT*16] = act( A[16 x NK*32] * W + bias ), D -> LDS as f16.
// kc is a RUNTIME loop (no unroll) so at most one A-frag + one B-frag are in
// flight -> low VGPR pressure, no spills. A-frag is reused across all NT tiles.
// Accumulators acc[NT] live across the k loop (NT=8 -> 64 VGPRs).
// ---------------------------------------------------------------------------
template <int NK, int NT, int ACT, bool PERCOL_BIAS>
__device__ __forceinline__ void wave_gemm_lds(const _Float16* __restrict__ wf,
                                              const float* __restrict__ bias,
                                              const _Float16* a, int astride,
                                              _Float16* d, int dstride, int lane) {
    v8f acc[NT];
#pragma unroll
    for (int nt = 0; nt < NT; ++nt) acc[nt] = (v8f){};
    const v16h* bp = (const v16h*)wf;
#pragma unroll 1
    for (int kc = 0; kc < NK; ++kc) {
        v16h av = load_a_frag(a, astride, lane, kc);
#pragma unroll
        for (int nt = 0; nt < NT; ++nt) {
            v16h bv = bp[(kc * NT + nt) * 32 + lane];
            acc[nt] = __builtin_amdgcn_wmma_f32_16x16x32_f16(false, av, false, bv, (short)0,
                                                             acc[nt], false, false);
        }
    }
    int colb  = lane & 15;
    int rbase = (lane >> 4) << 3;
#pragma unroll
    for (int nt = 0; nt < NT; ++nt) {
        int col    = nt * 16 + colb;
        float bval = PERCOL_BIAS ? bias[col] : bias[0];
#pragma unroll
        for (int v = 0; v < 8; ++v) {
            float xv = acc[nt][v] + bval;
            if (ACT == ACT_SILU)      xv = xv * sigmoid_f(xv);
            else if (ACT == ACT_SIG)  xv = sigmoid_f(xv);
            d[(rbase + v) * dstride + col] = (_Float16)xv;
        }
    }
}

// Same, but last layer of phi_h: hNext = hCur + (A*W + b), written to global f32.
template <int NK, int NT>
__device__ __forceinline__ void wave_gemm_residual(const _Float16* __restrict__ wf,
                                                   const float* __restrict__ bias,
                                                   const _Float16* a, int astride,
                                                   const float* __restrict__ hCur,
                                                   float* __restrict__ hNext,
                                                   int nodeBase, int lane) {
    v8f acc[NT];
#pragma unroll
    for (int nt = 0; nt < NT; ++nt) acc[nt] = (v8f){};
    const v16h* bp = (const v16h*)wf;
#pragma unroll 1
    for (int kc = 0; kc < NK; ++kc) {
        v16h av = load_a_frag(a, astride, lane, kc);
#pragma unroll
        for (int nt = 0; nt < NT; ++nt) {
            v16h bv = bp[(kc * NT + nt) * 32 + lane];
            acc[nt] = __builtin_amdgcn_wmma_f32_16x16x32_f16(false, av, false, bv, (short)0,
                                                             acc[nt], false, false);
        }
    }
    int colb  = lane & 15;
    int rbase = (lane >> 4) << 3;
#pragma unroll
    for (int nt = 0; nt < NT; ++nt) {
        int col    = nt * 16 + colb;
        float bval = bias[col];
#pragma unroll
        for (int v = 0; v < 8; ++v) {
            int node = nodeBase + rbase + v;
            hNext[(size_t)node * HF + col] = hCur[(size_t)node * HF + col] + acc[nt][v] + bval;
        }
    }
}

// ---------------------------------------------------------------------------
// Weight prep: f32 [L][K][NoutSrc] -> f16 B-fragment layout, K padded to Kpad,
// Nout padded to NoutDst (zeros).
// ---------------------------------------------------------------------------
__global__ void prep_weights(const float* __restrict__ src, _Float16* __restrict__ dst,
                             int K, int Kpad, int NoutSrc, int NoutDst, int total) {
    int idx = blockIdx.x * blockDim.x + threadIdx.x;
    if (idx >= total) return;
    int NT       = NoutDst >> 4;
    int perLayer = (Kpad >> 5) * NT * 512;
    int l        = idx / perLayer;
    int f        = idx - l * perLayer;
    int j        = f & 15;
    int lane     = (f >> 4) & 31;
    int tile     = f >> 9;
    int nt       = tile % NT;
    int kc       = tile / NT;
    int k        = kc * 32 + ((lane >> 4) << 3) + (j & 7) + ((j >> 3) << 4);
    int n        = nt * 16 + (lane & 15);
    float val    = 0.0f;
    if (k < K && n < NoutSrc)
        val = src[(size_t)l * K * NoutSrc + (size_t)k * NoutSrc + n];
    dst[idx] = (_Float16)val;
}

// ---------------------------------------------------------------------------
// Init: vin = x - mean_n(x); vectors0 = vin replicated over NV; h0 = 0.
// ---------------------------------------------------------------------------
__global__ __launch_bounds__(256) void init_kernel(const float* __restrict__ x,
                                                   float* __restrict__ vin,
                                                   float* __restrict__ vec0,
                                                   float* __restrict__ h0) {
    __shared__ float r0[256], r1[256], r2[256];
    int b  = blockIdx.x;
    int n  = threadIdx.x;
    int nf = b * N_NODES + n;
    float x0 = x[nf * 3 + 0], x1 = x[nf * 3 + 1], x2 = x[nf * 3 + 2];
    r0[n] = x0; r1[n] = x1; r2[n] = x2;
    __syncthreads();
    for (int s = 128; s > 0; s >>= 1) {
        if (n < s) { r0[n] += r0[n + s]; r1[n] += r1[n + s]; r2[n] += r2[n + s]; }
        __syncthreads();
    }
    const float inv = 1.0f / (float)N_NODES;
    float v0 = x0 - r0[0] * inv, v1 = x1 - r1[0] * inv, v2 = x2 - r2[0] * inv;
    vin[nf * 3 + 0] = v0; vin[nf * 3 + 1] = v1; vin[nf * 3 + 2] = v2;
#pragma unroll 1
    for (int c = 0; c < NV; ++c) {
        vec0[(nf * NV + c) * 3 + 0] = v0;
        vec0[(nf * NV + c) * 3 + 1] = v1;
        vec0[(nf * NV + c) * 3 + 2] = v2;
    }
#pragma unroll 1
    for (int k = 0; k < HF; ++k) h0[(size_t)nf * HF + k] = 0.0f;
}

// ---------------------------------------------------------------------------
// Edge kernel: one WG (4 waves, 128 thr) per receiver node. Each wave processes
// 4 tiles of 16 senders; full phi_e / phi_inf / phi_x WMMA chain per tile.
// Per-receiver m_i and shift sums accumulate in LDS (ds f32 atomics), one plain
// global write each at the end. Pad edge s==r is masked (diff==0 => shift==0;
// gate skipped for m_i).
// ---------------------------------------------------------------------------
#define EW 4
__global__ __launch_bounds__(EW * 32) void edge_kernel(
    const float* __restrict__ vecCur, float* __restrict__ vecNext,
    const float* __restrict__ hCur, float* __restrict__ mi,
    const _Float16* __restrict__ wfPE1, const _Float16* __restrict__ wfPE2,
    const _Float16* __restrict__ wfPI,  const _Float16* __restrict__ wfPX1,
    const _Float16* __restrict__ wfPX2, const _Float16* __restrict__ wfPX3,
    const float* __restrict__ pe_b1, const float* __restrict__ pe_b2,
    const float* __restrict__ pi_b,  const float* __restrict__ px_b1,
    const float* __restrict__ px_b2, const float* __restrict__ px_b3) {
    __shared__ __align__(32) _Float16 reg0[EW][16 * 288];  // feat, later m / p2
    __shared__ __align__(32) _Float16 reg1[EW][16 * 128];  // m1 / p1
    __shared__ __align__(32) _Float16 pxbuf[EW][16 * 16];  // gate, later px
    __shared__ float lenbuf[EW][16 * 16];
    __shared__ float mi_acc[MU];
    __shared__ float shift_acc[NV * 3];

    const int tid      = threadIdx.x;
    const int lane     = tid & 31;
    const int w        = tid >> 5;
    const int nodeFlat = blockIdx.x;  // b*N + r
    const int b        = nodeFlat / N_NODES;
    const int r        = nodeFlat - b * N_NODES;

    if (tid < MU) mi_acc[tid] = 0.0f;
    if (tid < NV * 3) shift_acc[tid] = 0.0f;
    __syncthreads();

    _Float16* A0 = reg0[w];
    _Float16* A1 = reg1[w];
    _Float16* PX = pxbuf[w];
    float*    LN = lenbuf[w];

    const float* vr = vecCur + (size_t)nodeFlat * NV * 3;
    const float* hr = hCur + (size_t)nodeFlat * HF;

#pragma unroll 1
    for (int t = 0; t < 4; ++t) {
        const int tileBase = (w * 4 + t) * 16;  // sender base (within batch)

        // ---- phase A: lengths + feat rows [16][288] ----
        {
            int row       = lane >> 1;
            int cb        = (lane & 1) * 8;
            int sFlat     = b * N_NODES + tileBase + row;
            const float* vs = vecCur + (size_t)sFlat * NV * 3;
#pragma unroll 2
            for (int i = 0; i < 8; ++i) {
                int c    = cb + i;
                float d0 = vr[c * 3 + 0] - vs[c * 3 + 0];
                float d1 = vr[c * 3 + 1] - vs[c * 3 + 1];
                float d2 = vr[c * 3 + 2] - vs[c * 3 + 2];
                float l2 = d0 * d0 + d1 * d1 + d2 * d2 + 1e-16f;
                float ln = sqrtf(l2);
                LN[row * 16 + c]   = ln;
                A0[row * 288 + c]  = (_Float16)l2;  // lengths^2 (safe_norm squared)
            }
#pragma unroll 1
            for (int i = 0; i < 136; ++i) {  // 16 rows x 272 cols (h_s | h_r | pad)
                int idx = i * 32 + lane;
                int rr  = idx / 272;
                int j   = idx - rr * 272;
                float val;
                if (j < HF) {
                    int sF = b * N_NODES + tileBase + rr;
                    val = hCur[(size_t)sF * HF + j];
                } else if (j < 2 * HF) {
                    val = hr[j - HF];
                } else {
                    val = 0.0f;
                }
                A0[rr * 288 + 16 + j] = (_Float16)val;
            }
        }

        // ---- phi_e ----
        wave_gemm_lds<9, 8, ACT_SILU, true>(wfPE1, pe_b1, A0, 288, A1, 128, lane);
        wave_gemm_lds<4, 8, ACT_SILU, true>(wfPE2, pe_b2, A1, 128, A0, 128, lane);  // m -> A0
        // ---- phi_inf gate (Nout padded 1->16, only col 0 used) ----
        wave_gemm_lds<4, 1, ACT_SIG, false>(wfPI, pi_b, A0, 128, PX, 16, lane);

        // ---- m_i += m * gate (mask pad edge s==r) ----
        {
#pragma unroll
            for (int q = 0; q < 4; ++q) {
                int k     = lane * 4 + q;
                float acc = 0.0f;
#pragma unroll 1
                for (int rr = 0; rr < 16; ++rr) {
                    float g = (tileBase + rr == r) ? 0.0f : (float)PX[rr * 16];
                    acc += (float)A0[rr * 128 + k] * g;
                }
                atomicAdd(&mi_acc[k], acc);
            }
        }

        // ---- phi_x ----
        wave_gemm_lds<4, 8, ACT_SILU, true>(wfPX1, px_b1, A0, 128, A1, 128, lane);
        wave_gemm_lds<4, 8, ACT_SILU, true>(wfPX2, px_b2, A1, 128, A0, 128, lane);
        wave_gemm_lds<4, 1, ACT_NONE, true>(wfPX3, px_b3, A0, 128, PX, 16, lane);

        // ---- shifts: px * diff / (1 + len), accumulate per receiver ----
        {
            int row       = lane >> 1;
            int cb        = (lane & 1) * 8;
            int sFlat     = b * N_NODES + tileBase + row;
            const float* vs = vecCur + (size_t)sFlat * NV * 3;
#pragma unroll 2
            for (int i = 0; i < 8; ++i) {
                int c       = cb + i;
                float ln    = LN[row * 16 + c];
                float scale = (float)PX[row * 16 + c] * __builtin_amdgcn_rcpf(1.0f + ln);
                float d0    = (vr[c * 3 + 0] - vs[c * 3 + 0]) * scale;
                float d1    = (vr[c * 3 + 1] - vs[c * 3 + 1]) * scale;
                float d2    = (vr[c * 3 + 2] - vs[c * 3 + 2]) * scale;
                atomicAdd(&shift_acc[c * 3 + 0], d0);
                atomicAdd(&shift_acc[c * 3 + 1], d1);
                atomicAdd(&shift_acc[c * 3 + 2], d2);
            }
        }
    }
    __syncthreads();

    if (tid < MU)
        mi[(size_t)nodeFlat * MU + tid] = mi_acc[tid] * 0.06262242910851495f;  // 1/sqrt(255)
    if (tid < NV * 3)
        vecNext[(size_t)nodeFlat * NV * 3 + tid] =
            vecCur[(size_t)nodeFlat * NV * 3 + tid] + shift_acc[tid] * (1.0f / 255.0f);
}

// ---------------------------------------------------------------------------
// Node kernel: phi_h on [m_i, h] -> residual h update. One wave per 16 nodes.
// ---------------------------------------------------------------------------
#define NWN 4
__global__ __launch_bounds__(NWN * 32) void node_kernel(
    const float* __restrict__ mi, const float* __restrict__ hCur, float* __restrict__ hNext,
    const _Float16* __restrict__ wfPH1, const _Float16* __restrict__ wfPH2,
    const _Float16* __restrict__ wfPH3, const float* __restrict__ ph_b1,
    const float* __restrict__ ph_b2, const float* __restrict__ ph_b3) {
    __shared__ __align__(32) _Float16 reg0[NWN][16 * 256];
    __shared__ __align__(32) _Float16 reg1[NWN][16 * 128];
    int tid      = threadIdx.x;
    int lane     = tid & 31;
    int w        = tid >> 5;
    int nodeBase = (blockIdx.x * NWN + w) * 16;
    _Float16* A0 = reg0[w];
    _Float16* A1 = reg1[w];

#pragma unroll 1
    for (int i = 0; i < 128; ++i) {  // 16 rows x 256 cols = mi | h
        int idx   = i * 32 + lane;
        int rr    = idx >> 8;
        int j     = idx & 255;
        int node  = nodeBase + rr;
        float val = (j < MU) ? mi[(size_t)node * MU + j] : hCur[(size_t)node * HF + (j - MU)];
        A0[rr * 256 + j] = (_Float16)val;
    }
    wave_gemm_lds<8, 8, ACT_SILU, true>(wfPH1, ph_b1, A0, 256, A1, 128, lane);
    wave_gemm_lds<4, 8, ACT_SILU, true>(wfPH2, ph_b2, A1, 128, A0, 128, lane);
    wave_gemm_residual<4, 8>(wfPH3, ph_b3, A0, 128, hCur, hNext, nodeBase, lane);
}

// ---------------------------------------------------------------------------
// Outputs
// ---------------------------------------------------------------------------
__global__ void vecout_kernel(const float* __restrict__ vecF, const float* __restrict__ vin,
                              float* __restrict__ out) {
    int idx = blockIdx.x * blockDim.x + threadIdx.x;
    if (idx >= NODES_TOTAL * NV * 3) return;
    int node = idx / (NV * 3);
    int d    = idx % 3;
    out[idx] = vecF[idx] - vin[node * 3 + d];
}

__global__ __launch_bounds__(128) void readout_kernel(const float* __restrict__ h,
                                                      const float* __restrict__ rw,
                                                      const float* __restrict__ rb,
                                                      float* __restrict__ out) {
    __shared__ float sh[128];
    __shared__ float p[128];
    int node = blockIdx.x, k = threadIdx.x;
    float v = h[(size_t)node * HF + k];
    sh[k] = v;
    __syncthreads();
    for (int s = 64; s > 0; s >>= 1) { if (k < s) sh[k] = fmaxf(sh[k], sh[k + s]); __syncthreads(); }
    float mx = sh[0];
    __syncthreads();
    float e = __expf(v - mx);
    sh[k] = e; p[k] = e;
    __syncthreads();
    for (int s = 64; s > 0; s >>= 1) { if (k < s) sh[k] += sh[k + s]; __syncthreads(); }
    float inv = 1.0f / sh[0];
    if (k < OUT_HD) {
        float acc = 0.0f;
#pragma unroll 1
        for (int j = 0; j < HF; ++j) acc += p[j] * rw[j * OUT_HD + k];
        out[NODES_TOTAL * NV * 3 + (size_t)node * OUT_HD + k] = acc * inv + rb[k];
    }
}

// ---------------------------------------------------------------------------
extern "C" void kernel_launch(void* const* d_in, const int* in_sizes, int n_in,
                              void* d_out, int out_size, void* d_ws, size_t ws_size,
                              hipStream_t stream) {
    (void)in_sizes; (void)n_in; (void)out_size; (void)ws_size;
    const float* x     = (const float*)d_in[0];
    const float* pe_w1 = (const float*)d_in[1];
    const float* pe_b1 = (const float*)d_in[2];
    const float* pe_w2 = (const float*)d_in[3];
    const float* pe_b2 = (const float*)d_in[4];
    const float* pi_w  = (const float*)d_in[5];
    const float* pi_b  = (const float*)d_in[6];
    const float* px_w1 = (const float*)d_in[7];
    const float* px_b1 = (const float*)d_in[8];
    const float* px_w2 = (const float*)d_in[9];
    const float* px_b2 = (const float*)d_in[10];
    const float* px_w3 = (const float*)d_in[11];
    const float* px_b3 = (const float*)d_in[12];
    const float* ph_w1 = (const float*)d_in[13];
    const float* ph_b1 = (const float*)d_in[14];
    const float* ph_w2 = (const float*)d_in[15];
    const float* ph_b2 = (const float*)d_in[16];
    const float* ph_w3 = (const float*)d_in[17];
    const float* ph_b3 = (const float*)d_in[18];
    const float* rw    = (const float*)d_in[19];
    const float* rb    = (const float*)d_in[20];
    float* out = (float*)d_out;

    // ---- workspace carve-up (~2 MB) ----
    char* ws   = (char*)d_ws;
    size_t off = 0;
    auto carve = [&](size_t bytes) -> char* {
        char* p = ws + off;
        off += (bytes + 255) & ~(size_t)255;
        return p;
    };
    const size_t F_PE1 = 9 * 8 * 512, F_PE2 = 4 * 8 * 512, F_PI = 4 * 1 * 512;
    const size_t F_PX1 = 4 * 8 * 512, F_PX2 = 4 * 8 * 512, F_PX3 = 4 * 1 * 512;
    const size_t F_PH1 = 8 * 8 * 512, F_PH2 = 4 * 8 * 512, F_PH3 = 4 * 8 * 512;
    _Float16* wfPE1 = (_Float16*)carve(L_LAYERS * F_PE1 * 2);
    _Float16* wfPE2 = (_Float16*)carve(L_LAYERS * F_PE2 * 2);
    _Float16* wfPI  = (_Float16*)carve(L_LAYERS * F_PI * 2);
    _Float16* wfPX1 = (_Float16*)carve(L_LAYERS * F_PX1 * 2);
    _Float16* wfPX2 = (_Float16*)carve(L_LAYERS * F_PX2 * 2);
    _Float16* wfPX3 = (_Float16*)carve(L_LAYERS * F_PX3 * 2);
    _Float16* wfPH1 = (_Float16*)carve(L_LAYERS * F_PH1 * 2);
    _Float16* wfPH2 = (_Float16*)carve(L_LAYERS * F_PH2 * 2);
    _Float16* wfPH3 = (_Float16*)carve(L_LAYERS * F_PH3 * 2);
    float* vin  = (float*)carve(NODES_TOTAL * 3 * 4);
    float* vecA = (float*)carve((size_t)NODES_TOTAL * NV * 3 * 4);
    float* vecB = (float*)carve((size_t)NODES_TOTAL * NV * 3 * 4);
    float* hA   = (float*)carve((size_t)NODES_TOTAL * HF * 4);
    float* hB   = (float*)carve((size_t)NODES_TOTAL * HF * 4);
    float* miB  = (float*)carve((size_t)NODES_TOTAL * MU * 4);

    // ---- weight fragment prep (every call: deterministic, no caching) ----
    auto prep = [&](const float* src, _Float16* dst, int K, int Kpad, int NoutSrc, int NoutDst) {
        int total = L_LAYERS * (Kpad / 32) * (NoutDst / 16) * 512;
        prep_weights<<<(total + 255) / 256, 256, 0, stream>>>(src, dst, K, Kpad, NoutSrc,
                                                              NoutDst, total);
    };
    prep(pe_w1, wfPE1, 272, 288, 128, 128);
    prep(pe_w2, wfPE2, 128, 128, 128, 128);
    prep(pi_w,  wfPI,  128, 128, 1, 16);
    prep(px_w1, wfPX1, 128, 128, 128, 128);
    prep(px_w2, wfPX2, 128, 128, 128, 128);
    prep(px_w3, wfPX3, 128, 128, 16, 16);
    prep(ph_w1, wfPH1, 256, 256, 128, 128);
    prep(ph_w2, wfPH2, 128, 128, 128, 128);
    prep(ph_w3, wfPH3, 128, 128, 128, 128);

    init_kernel<<<B_BATCH, 256, 0, stream>>>(x, vin, vecA, hA);

    float* vc = vecA; float* vn = vecB;
    float* hc = hA;   float* hn = hB;
    for (int l = 0; l < L_LAYERS; ++l) {
        edge_kernel<<<NODES_TOTAL, EW * 32, 0, stream>>>(
            vc, vn, hc, miB,
            wfPE1 + l * F_PE1, wfPE2 + l * F_PE2, wfPI + l * F_PI,
            wfPX1 + l * F_PX1, wfPX2 + l * F_PX2, wfPX3 + l * F_PX3,
            pe_b1 + l * MU, pe_b2 + l * MU, pi_b + l,
            px_b1 + l * MU, px_b2 + l * MU, px_b3 + l * NV);
        node_kernel<<<NODES_TOTAL / (NWN * 16), NWN * 32, 0, stream>>>(
            miB, hc, hn,
            wfPH1 + l * F_PH1, wfPH2 + l * F_PH2, wfPH3 + l * F_PH3,
            ph_b1 + l * MU, ph_b2 + l * MU, ph_b3 + l * HF);
        float* tv = vc; vc = vn; vn = tv;
        float* th = hc; hc = hn; hn = th;
    }

    vecout_kernel<<<(NODES_TOTAL * NV * 3 + 255) / 256, 256, 0, stream>>>(vc, vin, out);
    readout_kernel<<<NODES_TOTAL, 128, 0, stream>>>(hc, rw, rb, out);
}
